// SelfAttention_11373073400357
// MI455X (gfx1250) — matrix operand order repaired and checked by reading, not compiled, not run on hardware
//
#include <hip/hip_runtime.h>
#include <hip/hip_bf16.h>

// ---------------------------------------------------------------------------
// Self-attention forward, CDNA5 (gfx1250), wave32 WMMA bf16 pipeline with
// LDS staging via async-to-LDS loads (ASYNCcnt) and the Tensor Data Mover
// (TENSORcnt).  B=4, N=2048, C=1024, H=16, Dh=64.
// ---------------------------------------------------------------------------

#define B_   4
#define N_   2048
#define C_   1024
#define H_   16
#define DH_  64
#define M_   (B_ * N_)        // 8192 rows
#define QKVC (3 * C_)         // 3072

typedef __attribute__((ext_vector_type(16))) __bf16        v16bf;
typedef __attribute__((ext_vector_type(8)))  float         v8f;
typedef __attribute__((ext_vector_type(4)))  unsigned int  u32x4;
typedef __attribute__((ext_vector_type(8)))  int           i32x8;
typedef __attribute__((ext_vector_type(4)))  int           i32x4;
typedef int gv4i __attribute__((vector_size(16)));         // matches builtin param

union BF16Frag {
    v16bf          v;
    unsigned short u[16];
    uint4          q[2];
};

static __device__ __forceinline__ unsigned short f2bf(float f) {
    union { float f; unsigned u; } x{f};
    unsigned u = x.u;
    u += 0x7FFFu + ((u >> 16) & 1u);   // round-to-nearest-even
    return (unsigned short)(u >> 16);
}

static __device__ __forceinline__ v8f wmma_bf16(const BF16Frag& a, const BF16Frag& b, v8f c) {
    return __builtin_amdgcn_wmma_f32_16x16x32_bf16(
        /*neg_a=*/false, a.v, /*neg_b=*/false, b.v,
        /*c_mod=*/(short)0, c, /*reuse_a=*/false, /*reuse_b=*/false);
}

// ------------------- CDNA5 async / tensor data movement --------------------

#if __has_builtin(__builtin_amdgcn_global_load_async_to_lds_b128)
#define HAVE_ASYNC_LDS 1
static __device__ __forceinline__ void async_copy16(const unsigned short* g, unsigned short* l) {
    __builtin_amdgcn_global_load_async_to_lds_b128(
        (__attribute__((address_space(1))) gv4i*)g,
        (__attribute__((address_space(3))) gv4i*)l,
        /*offset=*/0, /*cpol=*/0);
}
static __device__ __forceinline__ void wait_async() {
#if __has_builtin(__builtin_amdgcn_s_wait_asynccnt)
    __builtin_amdgcn_s_wait_asynccnt(0);
#else
    asm volatile("s_wait_asynccnt 0" ::: "memory");
#endif
}
#else
#define HAVE_ASYNC_LDS 0
static __device__ __forceinline__ void async_copy16(const unsigned short* g, unsigned short* l) {
    *(uint4*)l = *(const uint4*)g;     // fallback: vmem load + ds store
}
static __device__ __forceinline__ void wait_async() {}
#endif

#if __has_builtin(__builtin_amdgcn_tensor_load_to_lds)
#define HAVE_TDM 1
// 2D tile load via TDM.  Descriptor packing per CDNA5 ISA
// (D# group0 = 4 dwords, group1 = 8 dwords; groups 2/3 zero for 2D).
// pad_enable inserts pad_amount dwords into LDS every pad_interval dwords.
static __device__ __forceinline__ void tdm_load_2d(
    unsigned lds_byte_off, const void* gptr,
    unsigned tile_w, unsigned tile_h,             // elements / rows
    unsigned tensor_w, unsigned tensor_h,
    unsigned long long row_stride_elems,
    unsigned pad_interval_code, unsigned pad_amount_code)
{
    unsigned long long ga = (unsigned long long)gptr;
    u32x4 g0;
    g0[0] = 1u;                                            // count=1, user D#
    g0[1] = lds_byte_off;                                  // lds_addr
    g0[2] = (unsigned)(ga & 0xffffffffu);                  // global_addr[31:0]
    g0[3] = (unsigned)((ga >> 32) & 0x01ffffffu)           // global_addr[56:32]
          | (2u << 30);                                    // type = 2 ("image")
    i32x8 g1;
    unsigned d0 = (1u << 16)                               // data_size = 2 bytes
                | (1u << 20)                               // pad_enable
                | (pad_interval_code << 22)
                | (pad_amount_code << 25);
    g1[0] = (int)d0;
    g1[1] = (int)((tensor_w & 0xffffu) << 16);             // tensor_dim0[15:0]
    g1[2] = (int)(((tensor_w >> 16) & 0xffffu) | ((tensor_h & 0xffffu) << 16));
    g1[3] = (int)(((tensor_h >> 16) & 0xffffu) | ((tile_w & 0xffffu) << 16));
    g1[4] = (int)(tile_h & 0xffffu);                       // tile_dim1, tile_dim2=0
    g1[5] = (int)(row_stride_elems & 0xffffffffu);         // tensor_dim0_stride lo
    g1[6] = (int)((row_stride_elems >> 32) & 0xffffu);     // stride hi, dim1_stride=0
    g1[7] = 0;
    i32x4 z4 = {0, 0, 0, 0};
#if defined(__clang_major__) && (__clang_major__ >= 23)
    i32x8 z8 = {0, 0, 0, 0, 0, 0, 0, 0};
    __builtin_amdgcn_tensor_load_to_lds(g0, g1, z4, z4, z8, 0);
#else
    __builtin_amdgcn_tensor_load_to_lds(g0, g1, z4, z4, 0);
#endif
}
static __device__ __forceinline__ void wait_tensor() {
#if __has_builtin(__builtin_amdgcn_s_wait_tensorcnt)
    __builtin_amdgcn_s_wait_tensorcnt(0);
#else
    asm volatile("s_wait_tensorcnt 0" ::: "memory");
#endif
}
#else
#define HAVE_TDM 0
#endif

// --------------------------- conversion kernels ----------------------------

__global__ void cvt_f32_bf16(const float* __restrict__ src, unsigned short* __restrict__ dst, int n) {
    int i = blockIdx.x * blockDim.x + threadIdx.x;
    int stride = gridDim.x * blockDim.x;
    for (; i < n; i += stride) dst[i] = f2bf(src[i]);
}

// src[rows][cols] (f32) -> dst[cols][rows] (bf16)
__global__ void cvt_transpose(const float* __restrict__ src, unsigned short* __restrict__ dst,
                              int rows, int cols) {
    int i = blockIdx.x * blockDim.x + threadIdx.x;
    int n = rows * cols;
    int stride = gridDim.x * blockDim.x;
    for (; i < n; i += stride) {
        int r = i / cols, c = i - r * cols;
        dst[c * rows + r] = f2bf(src[i]);
    }
}

// ------------------------ shared GEMM main loop ----------------------------
// WG = 256 threads = 8 waves; WG tile = 128(M) x 64(N).  The 64x32 B panel is
// staged into LDS (double buffered, rows padded to 40 ushorts) with async-to-
// LDS loads; all 8 waves consume it (8x reuse).  A fragments come straight
// from global (b128, L2-resident).

#define BP_ROW 40                     // padded LDS row: 32 bf16 + 8 pad
#define BP_ELEMS (64 * BP_ROW)

static __device__ __forceinline__ void gemm_mainloop(
    const unsigned short* __restrict__ arow,      // this lane's A row (m0+ln)
    const unsigned short* __restrict__ Bt,        // [Ncols][K] bf16
    int K, int n0, unsigned short* bp,            // LDS, 2*BP_ELEMS
    int tid, int hi, int ln, v8f acc[4])
{
    {   // prologue: stage k-block 0 into buffer 0
        int row = tid >> 2, ch = tid & 3;
        async_copy16(Bt + (size_t)(n0 + row) * K + ch * 8, bp + row * BP_ROW + ch * 8);
    }
    for (int kb = 0; kb < K; kb += 32) {
        unsigned short* cur = bp + ((kb >> 5) & 1) * BP_ELEMS;
        unsigned short* nxt = bp + ((((kb >> 5) & 1)) ^ 1) * BP_ELEMS;
        wait_async();
        __syncthreads();
        if (kb + 32 < K) {
            int row = tid >> 2, ch = tid & 3;
            async_copy16(Bt + (size_t)(n0 + row) * K + kb + 32 + ch * 8,
                         nxt + row * BP_ROW + ch * 8);
        }
        __builtin_prefetch(arow + kb + 256, 0, 1);
        BF16Frag a;
        a.q[0] = *(const uint4*)(arow + kb + 16 * hi);      // A elems 0..7
        a.q[1] = *(const uint4*)(arow + kb + 16 * hi + 8);  // A elems 8..15
        for (int t = 0; t < 4; ++t) {
            const unsigned short* bl = cur + (16 * t + ln) * BP_ROW + 16 * hi;
            BF16Frag bfr;
            bfr.q[0] = *(const uint4*)(bl);
            bfr.q[1] = *(const uint4*)(bl + 8);
            acc[t] = wmma_bf16(a, bfr, acc[t]);
        }
    }
}

// ------------------------------ QKV GEMM -----------------------------------

__global__ void __launch_bounds__(256) qkv_gemm(
        const unsigned short* __restrict__ xb,      // [8192][1024]
        const unsigned short* __restrict__ wqkvT,   // [3072][1024]
        unsigned short* __restrict__ Qd,
        unsigned short* __restrict__ Kd,
        unsigned short* __restrict__ Vt) {
    __shared__ __align__(16) unsigned short bp[2 * BP_ELEMS];
    const int tid = threadIdx.x;
    const int lane = tid & 31, wid = tid >> 5;
    const int hi = lane >> 4, ln = lane & 15;
    const int NT = QKVC / 64;                       // 48 column tiles
    const int mt = blockIdx.x / NT, nt = blockIdx.x - mt * NT;
    const int m0 = mt * 128 + wid * 16, n0 = nt * 64;

    v8f acc[4];
    for (int t = 0; t < 4; ++t) acc[t] = {};
    gemm_mainloop(xb + (size_t)(m0 + ln) * C_, wqkvT, C_, n0, bp, tid, hi, ln, acc);

    for (int t = 0; t < 4; ++t) {
        int n = n0 + 16 * t + ln;
        int which = n >> 10;                        // 0=Q 1=K 2=V
        int rem = n & 1023;
        int h = rem >> 6, d = rem & 63;
        for (int r = 0; r < 8; ++r) {
            int m = m0 + r + 8 * hi;
            int b = m >> 11, nn = m & 2047;
            unsigned short val = f2bf(acc[t][r]);
            if (which == 0)      Qd[(((b * H_ + h) * N_) + nn) * DH_ + d] = val;
            else if (which == 1) Kd[(((b * H_ + h) * N_) + nn) * DH_ + d] = val;
            else                 Vt[((b * H_ + h) * DH_ + d) * N_ + nn] = val;
        }
    }
}

// --------------------------- flash attention -------------------------------
// Block = 8 waves covering 8 consecutive 16-query blocks of ONE (b,h); the
// 32-key K/V chunk is staged once into LDS (TDM if available) and consumed by
// all 8 waves.  S is computed transposed (S^T = Kc @ Q^T) so exp(S) lands
// directly in the A-fragment layout required by P@V.

#define KL_ROW 72                      // 64 bf16 + 8 pad
#define VL_ROW 40                      // 32 bf16 + 8 pad

__global__ void __launch_bounds__(256) attn_kernel(
        const unsigned short* __restrict__ Qd,
        const unsigned short* __restrict__ Kd,
        const unsigned short* __restrict__ Vt,
        unsigned short* __restrict__ ctx) {
    __shared__ __align__(16) unsigned short klds[32 * KL_ROW];
    __shared__ __align__(16) unsigned short vlds[64 * VL_ROW];
    const int tid = threadIdx.x;
    const int lane = tid & 31, wid = tid >> 5;
    const int hi = lane >> 4, ln = lane & 15;
    const int bh = blockIdx.x >> 4;                 // 64 (b,h) pairs
    const int b = bh >> 4, h = bh & 15;
    const int q0 = ((blockIdx.x & 15) * 8 + wid) * 16;

    const unsigned short* Qbh = Qd + (size_t)bh * N_ * DH_;
    const unsigned short* Kbh = Kd + (size_t)bh * N_ * DH_;
    const unsigned short* Vbh = Vt + (size_t)bh * DH_ * N_;

    // Q^T as two B-fragments (d = 0..31, 32..63); lane n = query = ln.
    BF16Frag qf[2];
    {
        const unsigned short* qrow = Qbh + (q0 + ln) * DH_;
        qf[0].q[0] = *(const uint4*)(qrow + 16 * hi);
        qf[0].q[1] = *(const uint4*)(qrow + 16 * hi + 8);
        qf[1].q[0] = *(const uint4*)(qrow + 32 + 16 * hi);
        qf[1].q[1] = *(const uint4*)(qrow + 32 + 16 * hi + 8);
    }

#if HAVE_TDM
    const unsigned klds_off = (unsigned)(size_t)(void*)klds;  // flat LDS addr low bits
    const unsigned vlds_off = (unsigned)(size_t)(void*)vlds;
#endif

    v8f o[4];
    for (int t = 0; t < 4; ++t) o[t] = {};
    float m_run = -1e30f, l_run = 0.f;
    const float scale = 0.125f;                     // Dh^-0.5

    for (int kc = 0; kc < N_; kc += 32) {
#if HAVE_TDM
        if (wid == 0) {
            // K chunk: 32 rows x 64 bf16 (row = 128B -> pad every 32 dwords)
            tdm_load_2d(klds_off, Kbh + (size_t)kc * DH_, DH_, 32, DH_, 32,
                        (unsigned long long)DH_, /*interval 32dw*/4, /*pad 4dw*/3);
            // V chunk: 64 rows x 32 bf16 (row = 64B -> pad every 16 dwords)
            tdm_load_2d(vlds_off, Vbh + kc, 32, DH_, 32, DH_,
                        (unsigned long long)N_, /*interval 16dw*/3, /*pad 4dw*/3);
            wait_tensor();
        }
#else
        {
            int kr = tid >> 3, kch = tid & 7;       // 32 rows x 8 chunks of 16B
            async_copy16(Kbh + (size_t)(kc + kr) * DH_ + kch * 8,
                         klds + kr * KL_ROW + kch * 8);
            int vr = tid >> 2, vch = tid & 3;       // 64 rows x 4 chunks of 16B
            async_copy16(Vbh + (size_t)vr * N_ + kc + vch * 8,
                         vlds + vr * VL_ROW + vch * 8);
            wait_async();
        }
#endif
        __syncthreads();

        // S^T tiles from LDS K: lane holds query=ln, key = kc + 16*j + r + 8*hi
        v8f st[2];
        for (int j = 0; j < 2; ++j) {
            const unsigned short* kl = klds + (16 * j + ln) * KL_ROW;
            BF16Frag ka0, ka1;
            ka0.q[0] = *(const uint4*)(kl + 16 * hi);
            ka0.q[1] = *(const uint4*)(kl + 16 * hi + 8);
            ka1.q[0] = *(const uint4*)(kl + 32 + 16 * hi);
            ka1.q[1] = *(const uint4*)(kl + 32 + 16 * hi + 8);
            v8f s = {};
            s = wmma_bf16(ka0, qf[0], s);
            s = wmma_bf16(ka1, qf[1], s);
            st[j] = s * scale;
        }
        float mx = st[0][0];
        for (int r = 1; r < 8; ++r) mx = fmaxf(mx, st[0][r]);
        for (int r = 0; r < 8; ++r) mx = fmaxf(mx, st[1][r]);
        mx = fmaxf(mx, __shfl_xor(mx, 16, 32));
        float m_new = fmaxf(m_run, mx);
        float alpha = __expf(m_run - m_new);

        float ar[8];
        for (int r = 0; r < 8; ++r) ar[r] = __shfl(alpha, 8 * hi + r, 32);
        for (int t = 0; t < 4; ++t)
            for (int r = 0; r < 8; ++r) o[t][r] *= ar[r];
        l_run *= alpha;

        BF16Frag pf;
        float rs = 0.f;
        for (int r = 0; r < 8; ++r) { float p = __expf(st[0][r] - m_new); rs += p; pf.u[r]     = f2bf(p); }
        for (int r = 0; r < 8; ++r) { float p = __expf(st[1][r] - m_new); rs += p; pf.u[8 + r] = f2bf(p); }
        rs += __shfl_xor(rs, 16, 32);
        l_run += rs;
        m_run = m_new;

        for (int t = 0; t < 4; ++t) {
            const unsigned short* vl = vlds + (16 * t + ln) * VL_ROW + 8 * hi;
            BF16Frag vb;
            vb.q[0] = *(const uint4*)(vl);
            vb.q[1] = *(const uint4*)(vl + 16);
            o[t] = wmma_bf16(pf, vb, o[t]);
        }
        __syncthreads();
    }

    float linv[8];
    for (int r = 0; r < 8; ++r) {
        float l = __shfl(l_run, 8 * hi + r, 32);
        linv[r] = 1.f / l;
    }
    for (int t = 0; t < 4; ++t) {
        int d = 16 * t + ln;
        for (int r = 0; r < 8; ++r) {
            int q = q0 + r + 8 * hi;
            ctx[(((size_t)b * N_ + q) * H_ + h) * DH_ + d] = f2bf(o[t][r] * linv[r]);
        }
    }
}

// --------------------------- output projection -----------------------------

__global__ void __launch_bounds__(256) out_gemm(
        const unsigned short* __restrict__ ctx,
        const unsigned short* __restrict__ woutT,   // [1024][1024]
        const float* __restrict__ b_out,
        float* __restrict__ out) {
    __shared__ __align__(16) unsigned short bp[2 * BP_ELEMS];
    const int tid = threadIdx.x;
    const int lane = tid & 31, wid = tid >> 5;
    const int hi = lane >> 4, ln = lane & 15;
    const int NT = C_ / 64;                         // 16 column tiles
    const int mt = blockIdx.x / NT, nt = blockIdx.x - mt * NT;
    const int m0 = mt * 128 + wid * 16, n0 = nt * 64;

    v8f acc[4];
    for (int t = 0; t < 4; ++t) acc[t] = {};
    gemm_mainloop(ctx + (size_t)(m0 + ln) * C_, woutT, C_, n0, bp, tid, hi, ln, acc);

    for (int t = 0; t < 4; ++t) {
        int n = n0 + 16 * t + ln;
        float bias = b_out[n];
        for (int r = 0; r < 8; ++r) {
            int m = m0 + r + 8 * hi;
            out[(size_t)m * C_ + n] = acc[t][r] + bias;
        }
    }
}

// ------------------------------- launcher ----------------------------------

extern "C" void kernel_launch(void* const* d_in, const int* in_sizes, int n_in,
                              void* d_out, int out_size, void* d_ws, size_t ws_size,
                              hipStream_t stream) {
    const float* x     = (const float*)d_in[0];   // [4,2048,1024]
    const float* W_qkv = (const float*)d_in[1];   // [1024,3072]
    const float* W_out = (const float*)d_in[2];   // [1024,1024]
    const float* b_out = (const float*)d_in[3];   // [1024]
    float* out = (float*)d_out;

    char* ws = (char*)d_ws;
    size_t off = 0;
    unsigned short* xb    = (unsigned short*)(ws + off); off += (size_t)M_ * C_ * 2;
    unsigned short* wqkvT = (unsigned short*)(ws + off); off += (size_t)QKVC * C_ * 2;
    unsigned short* woutT = (unsigned short*)(ws + off); off += (size_t)C_ * C_ * 2;
    unsigned short* Qd    = (unsigned short*)(ws + off); off += (size_t)M_ * C_ * 2;
    unsigned short* Kd    = (unsigned short*)(ws + off); off += (size_t)M_ * C_ * 2;
    unsigned short* Vt    = (unsigned short*)(ws + off); off += (size_t)M_ * C_ * 2;
    unsigned short* ctx   = (unsigned short*)(ws + off); off += (size_t)M_ * C_ * 2;
    (void)ws_size; (void)in_sizes; (void)n_in; (void)out_size;

    const int T = 256;

    cvt_f32_bf16 <<<4096, T, 0, stream>>>(x, xb, M_ * C_);
    cvt_transpose<<<2048, T, 0, stream>>>(W_qkv, wqkvT, C_, QKVC);
    cvt_transpose<<<1024, T, 0, stream>>>(W_out, woutT, C_, C_);

    // QKV projection: (8192/128) x (3072/64) workgroups
    qkv_gemm<<<(M_ / 128) * (QKVC / 64), T, 0, stream>>>(xb, wqkvT, Qd, Kd, Vt);

    // attention: 64 (b,h) x 16 blocks (8 query-tiles of 16 each)
    attn_kernel<<<B_ * H_ * (N_ / (16 * 8)), T, 0, stream>>>(Qd, Kd, Vt, ctx);

    // output projection: (8192/128) x (1024/64) workgroups
    out_gemm<<<(M_ / 128) * (C_ / 64), T, 0, stream>>>(ctx, woutT, b_out, out);
}
